// TaacMixedCausalAttention_5248450036132
// MI455X (gfx1250) — matrix operand order repaired and checked by reading, not compiled, not run on hardware
//
#include <hip/hip_runtime.h>
#include <stdint.h>

// Problem constants (match reference)
#define B_   4
#define S_   2048
#define D_   1024
#define H_   16
#define HD_  64
#define NS_  32
#define NSL_ 512
#define TQ_  544     // NSL + NS queries
#define TK_  2080    // S + NS keys
#define NEGV (-1.0e9f)

typedef __attribute__((ext_vector_type(16))) __bf16 bf16x16;
typedef __attribute__((ext_vector_type(8)))  float  f32x8;
typedef __attribute__((ext_vector_type(4)))  int    i32x4;

// typed address-space pointers for the async-LDS builtin
typedef __attribute__((address_space(1))) i32x4* gptr_b128;
typedef __attribute__((address_space(3))) i32x4* lptr_b128;

union Frag { unsigned int u[8]; bf16x16 v; };

#if __has_builtin(__builtin_amdgcn_global_load_async_to_lds_b128) && \
    __has_builtin(__builtin_amdgcn_s_wait_asynccnt)
#define USE_ASYNC_LDS 1
#else
#define USE_ASYNC_LDS 0
#endif

// Probe: hardware packed fp32x2 -> bf16x2 conversion (single VALU op).
// No clang builtin on this toolchain, so use the ISA mnemonic directly.
// Fallback (if assembler rejects): set USE_HW_CVT_BF16 to 0.
#define USE_HW_CVT_BF16 1

static __device__ __forceinline__ unsigned int pack_bf16(float a, float b) {
#if USE_HW_CVT_BF16
  unsigned int r;
  asm("v_cvt_pk_bf16_f32 %0, %1, %2" : "=v"(r) : "v"(a), "v"(b));
  return r;
#else
  unsigned int ua = __float_as_uint(a);
  ua += 0x7FFF + ((ua >> 16) & 1);
  unsigned int ub = __float_as_uint(b);
  ub += 0x7FFF + ((ub >> 16) & 1);
  return (ua >> 16) | (ub & 0xFFFF0000u);
#endif
}
static __device__ __forceinline__ unsigned short bf16s(float a) {
  return (unsigned short)pack_bf16(a, a);
}

static __device__ __forceinline__ f32x8 zero8() {
  f32x8 z = {0.f, 0.f, 0.f, 0.f, 0.f, 0.f, 0.f, 0.f};
  return z;
}

// ---------------------------------------------------------------------------
// bf16-WMMA GEMM:  C[M x 1024] = A[M x 1024] * W[1024 x 1024]
// Block tile 64x128, K-step 32, 8 waves (4 along M x 2 along N); each wave
// owns 16x64 = four 16x16 f32 accumulators.
// ---------------------------------------------------------------------------
template <int A_BF16, int FINAL>
__global__ __launch_bounds__(256) void gemm_wmma_kernel(
    const void* __restrict__ Avp, const float* __restrict__ W,
    unsigned short* __restrict__ Cb,            // FINAL==0 dest (bf16)
    float* __restrict__ outSeq,                 // FINAL==1 dests
    float* __restrict__ outNs,
    const unsigned char* __restrict__ seqMask,
    const unsigned char* __restrict__ nsMask,
    int M, int rowsPerBatch, int aBatchStride, int aRowOff,
    int cBatchRows, int cRowOff) {
  __shared__ unsigned short As[64 * 36];        // [m][k], stride 36 (8B align)
  __shared__ unsigned short Bs[128 * 36];       // [n][k] transposed W tile

  const int tid  = threadIdx.x;
  const int lane = tid & 31;
  const int wid  = tid >> 5;
  const int l15  = lane & 15;
  const int hi   = (lane >> 4) & 1;
  const int wm   = wid >> 1;                    // 0..3
  const int wn   = wid & 1;                     // 0..1
  const int mt   = blockIdx.x;
  const int nb   = blockIdx.y;                  // 8 tiles of 128 cols

  f32x8 acc[4];
#pragma unroll
  for (int t = 0; t < 4; ++t) acc[t] = zero8();

  // A staging: 4 threads per row, 8 contiguous elements each
  const int ar = tid >> 2;                      // 0..63
  const int ac = (tid & 3) << 3;                // 0,8,16,24
  long aIdx;
  {
    int row = mt * 64 + ar;
    int rr  = (row < M) ? row : 0;
    int rb  = rr / rowsPerBatch;
    int ri  = rr - rb * rowsPerBatch;
    aIdx = ((long)rb * aBatchStride + ri + aRowOff) * (long)D_;
  }
  // W staging: thread owns a k-pair x 8 columns (packed dword stores)
  const int kp = (tid >> 4) << 1;               // 0,2,..,30
  const int nc = (tid & 15) << 3;               // 0..120 step 8

  const float*          Af = (const float*)Avp;
  const unsigned short* Ab = (const unsigned short*)Avp;

  for (int k0 = 0; k0 < D_; k0 += 32) {
    __syncthreads();
    // ---- stage A (64x32) -> bf16 LDS (two uint2 stores) ----
    if (A_BF16) {
      uint4 x = *(const uint4*)(Ab + aIdx + k0 + ac);
      uint2 s0; s0.x = x.x; s0.y = x.y;
      uint2 s1; s1.x = x.z; s1.y = x.w;
      *(uint2*)&As[ar * 36 + ac]     = s0;
      *(uint2*)&As[ar * 36 + ac + 4] = s1;
    } else {
      const float4* g = (const float4*)(Af + aIdx + k0 + ac);
      float4 f0 = g[0], f1 = g[1];
      uint2 s0; s0.x = pack_bf16(f0.x, f0.y); s0.y = pack_bf16(f0.z, f0.w);
      uint2 s1; s1.x = pack_bf16(f1.x, f1.y); s1.y = pack_bf16(f1.z, f1.w);
      *(uint2*)&As[ar * 36 + ac]     = s0;
      *(uint2*)&As[ar * 36 + ac + 4] = s1;
    }
    // ---- stage W (32x128) -> transposed [n][k], k-pairs packed ----
    {
      const float* p0 = W + (long)(k0 + kp) * D_ + nb * 128 + nc;
      const float* p1 = p0 + D_;
      if (k0 + 32 < D_) __builtin_prefetch(p0 + 32 * (long)D_, 0, 0);
      const float4* q0 = (const float4*)p0;
      const float4* q1 = (const float4*)p1;
      float4 a0 = q0[0], a1 = q0[1], b0 = q1[0], b1 = q1[1];
      float fa[8] = {a0.x, a0.y, a0.z, a0.w, a1.x, a1.y, a1.z, a1.w};
      float fb[8] = {b0.x, b0.y, b0.z, b0.w, b1.x, b1.y, b1.z, b1.w};
#pragma unroll
      for (int i = 0; i < 8; ++i)
        *(unsigned int*)&Bs[(nc + i) * 36 + kp] = pack_bf16(fa[i], fb[i]);
    }
    __syncthreads();

    // A fragment (16x32): lane row = l15; K pairs per ISA layout
    Frag fa;
#pragma unroll
    for (int j = 0; j < 8; ++j) {
      int k = 2 * j + ((j >= 4) ? 8 : 0) + (hi ? 8 : 0);
      fa.u[j] = *(const unsigned int*)&As[(wm * 16 + l15) * 36 + k];
    }
#pragma unroll
    for (int t = 0; t < 4; ++t) {
      Frag fb;                                  // B fragment (32x16)
#pragma unroll
      for (int j = 0; j < 8; ++j) {
        int k = 2 * j + ((j >= 4) ? 8 : 0) + (hi ? 8 : 0);
        fb.u[j] = *(const unsigned int*)&Bs[(wn * 64 + t * 16 + l15) * 36 + k];
      }
      acc[t] = __builtin_amdgcn_wmma_f32_16x16x32_bf16(
          false, fa.v, false, fb.v, (short)0, acc[t], false, false);
    }
  }

  // epilogue: C layout -> row = wm*16 + r + 8*hi, col = lane&15 within tile
#pragma unroll
  for (int r = 0; r < 8; ++r) {
    int row = mt * 64 + wm * 16 + r + (hi ? 8 : 0);
    if (row >= M) continue;
    const int colb = nb * 128 + wn * 64 + l15;
    if (!FINAL) {
      int  rb = row / rowsPerBatch;
      int  ri = row - rb * rowsPerBatch;
      unsigned short* crow =
          Cb + ((long)rb * cBatchRows + ri + cRowOff) * D_ + colb;
#pragma unroll
      for (int t = 0; t < 4; ++t) crow[t * 16] = bf16s(acc[t][r]);
    } else {
      int b  = row / TQ_;
      int tq = row - b * TQ_;
      float  mv;
      float* drow;
      if (tq < NSL_) {
        mv   = (float)seqMask[b * S_ + (S_ - NSL_) + tq];
        drow = outSeq + ((long)(b * NSL_ + tq)) * D_ + colb;
      } else {
        int nn = tq - NSL_;
        mv   = (float)nsMask[b * NS_ + nn];
        drow = outNs + ((long)(b * NS_ + nn)) * D_ + colb;
      }
#pragma unroll
      for (int t = 0; t < 4; ++t) drow[t * 16] = acc[t][r] * mv;
    }
  }
}

// ---------------------------------------------------------------------------
// Per-ns-token projection: (4 rows of x) @ W_ns[n]; pure bandwidth -> GEMV.
// ---------------------------------------------------------------------------
__global__ __launch_bounds__(256) void ns_proj_kernel(
    const float* __restrict__ X,                 // (B, NS, D)
    const float* __restrict__ Wn,                // (NS, D, D)
    unsigned short* __restrict__ Cb, int cBatchRows, int cRowOff) {
  __shared__ float xs[B_ * D_];
  const int n   = blockIdx.x;
  const int col = blockIdx.y * 256 + threadIdx.x;
  for (int i = threadIdx.x; i < B_ * D_; i += 256) {
    int b = i >> 10, d = i & (D_ - 1);
    xs[i] = X[((long)b * NS_ + n) * D_ + d];
  }
  __syncthreads();
  float a0 = 0.f, a1 = 0.f, a2 = 0.f, a3 = 0.f;
  const float* w = Wn + (long)n * D_ * D_ + col;
#pragma unroll 8
  for (int d = 0; d < D_; ++d) {
    float wv = w[(long)d * D_];
    a0 += xs[d] * wv;
    a1 += xs[D_ + d] * wv;
    a2 += xs[2 * D_ + d] * wv;
    a3 += xs[3 * D_ + d] * wv;
  }
  const long r = cRowOff + n;
  Cb[(0L * cBatchRows + r) * D_ + col] = bf16s(a0);
  Cb[(1L * cBatchRows + r) * D_ + col] = bf16s(a1);
  Cb[(2L * cBatchRows + r) * D_ + col] = bf16s(a2);
  Cb[(3L * cBatchRows + r) * D_ + col] = bf16s(a3);
}

// ---------------------------------------------------------------------------
// Flash attention: grid (ceil(TQ/64), B*H), block 128 (4 waves x 16 queries).
// K tile staged via async-LDS b128; V staged transposed with packed key-pair
// dwords; scores & P*V via WMMA; online softmax with 16-lane shfl_xor row
// reductions; P restaged through per-wave LDS (A-frag layout).
// ---------------------------------------------------------------------------
__global__ __launch_bounds__(128) void attn_kernel(
    const unsigned short* __restrict__ Qb, const unsigned short* __restrict__ Kb,
    const unsigned short* __restrict__ Vb, unsigned short* __restrict__ Ob,
    const unsigned char* __restrict__ seqMask,
    const unsigned char* __restrict__ nsMask) {
  __shared__ unsigned short Ks[64 * 72];        // [key][d], 16B-aligned rows
  __shared__ unsigned short Vs[64 * 68];        // [d][key]  (transposed)
  __shared__ unsigned short Ps[4 * 16 * 68];    // per-wave P restage

  const int tid  = threadIdx.x;
  const int lane = tid & 31;
  const int wid  = tid >> 5;
  const int l15  = lane & 15;
  const int hi   = (lane >> 4) & 1;
  const int bh   = blockIdx.y;
  const int b    = bh >> 4;
  const int h    = bh & 15;
  const int qblk  = blockIdx.x * 64;
  const int qbase = qblk + wid * 16;

  // preload Q fragments for both 32-wide d-steps
  Frag aq[2];
  {
    int q = qbase + l15;
    if (q > TQ_ - 1) q = TQ_ - 1;
    const unsigned short* qp = Qb + ((long)(b * TQ_ + q)) * D_ + h * HD_;
#pragma unroll
    for (int s = 0; s < 2; ++s)
#pragma unroll
      for (int j = 0; j < 8; ++j) {
        int k = s * 32 + 2 * j + ((j >= 4) ? 8 : 0) + (hi ? 8 : 0);
        aq[s].u[j] = *(const unsigned int*)&qp[k];
      }
  }

  f32x8 ot[4];
#pragma unroll
  for (int t = 0; t < 4; ++t) ot[t] = zero8();
  float m_i[8], l_i[8];
#pragma unroll
  for (int r = 0; r < 8; ++r) { m_i[r] = -1.0e30f; l_i[r] = 0.f; }

  int qlast = qblk + 63;
  if (qlast > TQ_ - 1) qlast = TQ_ - 1;
  const int kmax = (S_ - NSL_) + qlast + 1;     // q_pos = 1536 + q

  // staging indices
  const int srow  = tid >> 1;                   // K: key row 0..63
  const int shalf = (tid & 1) * 32;             // K: d half
  const int vkp   = tid >> 2;                   // V: key pair 0..31
  const int vdq   = (tid & 3) * 16;             // V: d quarter

  for (int kb = 0; kb < kmax; kb += 64) {
    __syncthreads();
    // ---- stage K tile [key][d] ----
    {
      int kr = kb + srow;
      if (kr > TK_ - 1) kr = TK_ - 1;           // clamp; masked below
      unsigned short* gk = const_cast<unsigned short*>(
          Kb + ((long)(b * TK_ + kr)) * D_ + h * HD_ + shalf);
#if USE_ASYNC_LDS
#pragma unroll
      for (int i = 0; i < 4; ++i)
        __builtin_amdgcn_global_load_async_to_lds_b128(
            (gptr_b128)(gk + 8 * i),
            (lptr_b128)&Ks[srow * 72 + shalf + 8 * i], 0, 0);
#else
      const uint2* gs = (const uint2*)gk;
      uint2* ds = (uint2*)&Ks[srow * 72 + shalf];
#pragma unroll
      for (int i = 0; i < 8; ++i) ds[i] = gs[i];
#endif
    }
    // ---- stage V tile transposed [d][key], packed key pairs ----
    {
      int r0 = kb + 2 * vkp; if (r0 > TK_ - 1) r0 = TK_ - 1;
      int r1 = r0 + 1;       if (r1 > TK_ - 1) r1 = TK_ - 1;
      const uint4* v0 = (const uint4*)(Vb + ((long)(b * TK_ + r0)) * D_ + h * HD_ + vdq);
      const uint4* v1 = (const uint4*)(Vb + ((long)(b * TK_ + r1)) * D_ + h * HD_ + vdq);
      uint4 a0 = v0[0], a1 = v0[1], c0 = v1[0], c1 = v1[1];
      unsigned int av[8] = {a0.x, a0.y, a0.z, a0.w, a1.x, a1.y, a1.z, a1.w};
      unsigned int cv[8] = {c0.x, c0.y, c0.z, c0.w, c1.x, c1.y, c1.z, c1.w};
#pragma unroll
      for (int i = 0; i < 8; ++i) {
        unsigned int lo = (av[i] & 0xFFFFu) | (cv[i] << 16);
        unsigned int h2 = (av[i] >> 16) | (cv[i] & 0xFFFF0000u);
        *(unsigned int*)&Vs[(vdq + 2 * i) * 68 + 2 * vkp]     = lo;
        *(unsigned int*)&Vs[(vdq + 2 * i + 1) * 68 + 2 * vkp] = h2;
      }
    }
#if USE_ASYNC_LDS
    __builtin_amdgcn_s_wait_asynccnt(0);
#endif
    __syncthreads();

    // ---- S = Q * K^T (4 key tiles x 2 d-steps) ----
    f32x8 st[4];
#pragma unroll
    for (int t = 0; t < 4; ++t) st[t] = zero8();
#pragma unroll
    for (int s = 0; s < 2; ++s)
#pragma unroll
      for (int t = 0; t < 4; ++t) {
        Frag fb;
#pragma unroll
        for (int j = 0; j < 8; ++j) {
          int k = s * 32 + 2 * j + ((j >= 4) ? 8 : 0) + (hi ? 8 : 0);
          fb.u[j] = *(const unsigned int*)&Ks[(t * 16 + l15) * 72 + k];
        }
        st[t] = __builtin_amdgcn_wmma_f32_16x16x32_bf16(
            false, aq[s].v, false, fb.v, (short)0, st[t], false, false);
      }

    // ---- scale + causal/key masking ----
#pragma unroll
    for (int t = 0; t < 4; ++t) {
      const int kg = kb + t * 16 + l15;
      float km = 0.f;
      if (kg < TK_)
        km = (kg < S_) ? (float)seqMask[b * S_ + kg]
                       : (float)nsMask[b * NS_ + (kg - S_)];
#pragma unroll
      for (int r = 0; r < 8; ++r) {
        int  q    = qbase + r + (hi ? 8 : 0);
        bool ok   = (km != 0.f) && (kg <= (S_ - NSL_) + q) && (q < TQ_);
        st[t][r]  = ok ? st[t][r] * 0.125f : NEGV;
      }
    }

    // ---- online softmax ----
#pragma unroll
    for (int r = 0; r < 8; ++r) {
      float m = fmaxf(fmaxf(st[0][r], st[1][r]), fmaxf(st[2][r], st[3][r]));
#pragma unroll
      for (int off = 1; off < 16; off <<= 1) m = fmaxf(m, __shfl_xor(m, off, 32));
      float nm    = fmaxf(m_i[r], m);
      float alpha = __expf(m_i[r] - nm);
      m_i[r]      = nm;
      float ps    = 0.f;
#pragma unroll
      for (int t = 0; t < 4; ++t) {
        float p  = __expf(st[t][r] - nm);
        st[t][r] = p;
        ps += p;
      }
#pragma unroll
      for (int off = 1; off < 16; off <<= 1) ps += __shfl_xor(ps, off, 32);
      l_i[r] = l_i[r] * alpha + ps;
#pragma unroll
      for (int t = 0; t < 4; ++t) ot[t][r] *= alpha;
    }

    // ---- restage P (C layout -> A layout) through per-wave LDS ----
    unsigned short* pw = &Ps[wid * 16 * 68];
#pragma unroll
    for (int t = 0; t < 4; ++t)
#pragma unroll
      for (int r = 0; r < 8; ++r)
        pw[(r + (hi ? 8 : 0)) * 68 + t * 16 + l15] = bf16s(st[t][r]);

    // ---- O += P * V ----
#pragma unroll
    for (int s = 0; s < 2; ++s) {
      Frag ap;
#pragma unroll
      for (int j = 0; j < 8; ++j) {
        int k = s * 32 + 2 * j + ((j >= 4) ? 8 : 0) + (hi ? 8 : 0);
        ap.u[j] = *(const unsigned int*)&pw[l15 * 68 + k];
      }
#pragma unroll
      for (int t = 0; t < 4; ++t) {
        Frag bv;
#pragma unroll
        for (int j = 0; j < 8; ++j) {
          int k = s * 32 + 2 * j + ((j >= 4) ? 8 : 0) + (hi ? 8 : 0);
          bv.u[j] = *(const unsigned int*)&Vs[(t * 16 + l15) * 68 + k];
        }
        ot[t] = __builtin_amdgcn_wmma_f32_16x16x32_bf16(
            false, ap.v, false, bv.v, (short)0, ot[t], false, false);
      }
    }
  }

  // normalize & store attended (bf16)
#pragma unroll
  for (int r = 0; r < 8; ++r) {
    int q = qbase + r + (hi ? 8 : 0);
    if (q >= TQ_) continue;
    float inv = 1.f / fmaxf(l_i[r], 1.0e-30f);
    unsigned short* orow = Ob + ((long)(b * TQ_ + q)) * D_ + h * HD_ + l15;
#pragma unroll
    for (int t = 0; t < 4; ++t) orow[t * 16] = bf16s(ot[t][r] * inv);
  }
}

// Write seq_query_mask (middle output chunk) as float 0/1.
__global__ void mask_out_kernel(const unsigned char* __restrict__ seqMask,
                                float* __restrict__ outMask) {
  int i = blockIdx.x * 256 + threadIdx.x;
  if (i < B_ * NSL_) {
    int b = i / NSL_, t = i - b * NSL_;
    outMask[i] = (float)seqMask[b * S_ + (S_ - NSL_) + t];
  }
}

// ---------------------------------------------------------------------------
extern "C" void kernel_launch(void* const* d_in, const int* in_sizes, int n_in,
                              void* d_out, int out_size, void* d_ws,
                              size_t ws_size, hipStream_t stream) {
  (void)in_sizes; (void)n_in; (void)out_size; (void)ws_size;

  const float*         seqTok  = (const float*)d_in[0];
  const unsigned char* seqMask = (const unsigned char*)d_in[1];
  const float*         nsTok   = (const float*)d_in[2];
  const unsigned char* nsMask  = (const unsigned char*)d_in[3];
  // d_in[4] = next_sequence_length (compile-time NSL_ here)
  const float* Wq    = (const float*)d_in[5];
  const float* Wk    = (const float*)d_in[6];
  const float* Wv    = (const float*)d_in[7];
  const float* Wq_ns = (const float*)d_in[8];
  const float* Wk_ns = (const float*)d_in[9];
  const float* Wv_ns = (const float*)d_in[10];
  const float* Wo    = (const float*)d_in[11];

  // workspace: Q (B,544,D) | K (B,2080,D) | V (B,2080,D) | O (B,544,D), bf16
  unsigned short* Qb = (unsigned short*)d_ws;
  unsigned short* Kb = Qb + (size_t)B_ * TQ_ * D_;
  unsigned short* Vb = Kb + (size_t)B_ * TK_ * D_;
  unsigned short* Ob = Vb + (size_t)B_ * TK_ * D_;

  float* outSeq  = (float*)d_out;                          // (B,512,D)
  float* outMask = outSeq + (size_t)B_ * NSL_ * D_;        // (B,512)
  float* outNs   = outMask + (size_t)B_ * NSL_;            // (B,32,D)

  dim3 blk(256);

  // Q = seqTok[:, -512:] @ Wq  -> Qb rows [0,512) per batch
  gemm_wmma_kernel<0, 0><<<dim3((B_ * NSL_) / 64, 8), blk, 0, stream>>>(
      seqTok, Wq, Qb, nullptr, nullptr, nullptr, nullptr,
      B_ * NSL_, NSL_, S_, S_ - NSL_, TQ_, 0);
  // K = seqTok @ Wk -> Kb rows [0,2048) per batch
  gemm_wmma_kernel<0, 0><<<dim3((B_ * S_) / 64, 8), blk, 0, stream>>>(
      seqTok, Wk, Kb, nullptr, nullptr, nullptr, nullptr,
      B_ * S_, S_, S_, 0, TK_, 0);
  // V = seqTok @ Wv -> Vb rows [0,2048) per batch
  gemm_wmma_kernel<0, 0><<<dim3((B_ * S_) / 64, 8), blk, 0, stream>>>(
      seqTok, Wv, Vb, nullptr, nullptr, nullptr, nullptr,
      B_ * S_, S_, S_, 0, TK_, 0);

  // ns projections (rows 512..543 of Q, rows 2048..2079 of K/V)
  ns_proj_kernel<<<dim3(NS_, D_ / 256), blk, 0, stream>>>(nsTok, Wq_ns, Qb, TQ_, NSL_);
  ns_proj_kernel<<<dim3(NS_, D_ / 256), blk, 0, stream>>>(nsTok, Wk_ns, Kb, TK_, S_);
  ns_proj_kernel<<<dim3(NS_, D_ / 256), blk, 0, stream>>>(nsTok, Wv_ns, Vb, TK_, S_);

  // attention
  attn_kernel<<<dim3((TQ_ + 63) / 64, B_ * H_), dim3(128), 0, stream>>>(
      Qb, Kb, Vb, Ob, seqMask, nsMask);

  // out = attended @ Wo, masked + scattered into seq/ns output regions
  gemm_wmma_kernel<1, 1><<<dim3((B_ * TQ_) / 64, 8), blk, 0, stream>>>(
      Ob, Wo, nullptr, outSeq, outNs, seqMask, nsMask,
      B_ * TQ_, TQ_, TQ_, 0, 0, 0);

  // middle output chunk: seq_query_mask
  mask_out_kernel<<<dim3((B_ * NSL_ + 255) / 256), blk, 0, stream>>>(seqMask, outMask);
}
